// AutoCorrelation_56770877718639
// MI455X (gfx1250) — matrix operand order repaired and checked by reading, not compiled, not run on hardware
//
#include <hip/hip_runtime.h>
#include <hip/hip_bf16.h>
#include <stdint.h>

// AutoCorrelation (Autoformer) for MI455X / gfx1250.
// corr[t] = sum_n q[n]*k[(n-t) mod 512] computed as 32 dense 16x16x32 f16 WMMAs
// per row via 16-phase decimation (full matrix-core utilization), then top-7
// mean + sigmoid blend of value with time-rolled value.

typedef __attribute__((ext_vector_type(16))) _Float16 v16h;
typedef __attribute__((ext_vector_type(8)))  float    v8f;

#define DM   512
#define LSEQ 2048
#define WPB  8      // waves (rows) per block

__device__ __forceinline__ uint32_t f2h(float f) {
  _Float16 h = (_Float16)f;
  return (uint32_t)__builtin_bit_cast(unsigned short, h);
}

__global__ __launch_bounds__(256)
void autocorr_wmma_kernel(const float* __restrict__ Q,
                          const float* __restrict__ K,
                          const float* __restrict__ V,
                          float* __restrict__ Out) {
  // per-wave LDS: q phases (16 rows x 32 m, stride 34 halves, padded),
  // k phases (32 m-rows x 16 w, stride 18 halves -> conflict-free B gather),
  // corr accumulator (512 f32).
  __shared__ unsigned short qsh[WPB][16 * 34];
  __shared__ unsigned short ksh[WPB][32 * 18];
  __shared__ float          corr[WPB][DM];

  const int wave = threadIdx.x >> 5;
  const int lane = threadIdx.x & 31;
  const int row  = blockIdx.x * WPB + wave;
  const int lpos = row & (LSEQ - 1);
  const int rowN = (lpos == LSEQ - 1) ? row - (LSEQ - 1) : row + 1;  // roll(-1)

  const float* qrow = Q + (size_t)row * DM;
  const float* krow = K + (size_t)row * DM;

  // ---------------- stage q,k as f16 into LDS (coalesced 64B/lane loads) ----
  {
    float qv[16], kv[16];
    const float4* q4 = (const float4*)(qrow + (lane << 4));
    const float4* k4 = (const float4*)(krow + (lane << 4));
#pragma unroll
    for (int t = 0; t < 4; ++t) {
      float4 a = q4[t], b = k4[t];
      qv[4*t+0]=a.x; qv[4*t+1]=a.y; qv[4*t+2]=a.z; qv[4*t+3]=a.w;
      kv[4*t+0]=b.x; kv[4*t+1]=b.y; kv[4*t+2]=b.z; kv[4*t+3]=b.w;
    }
    // element idx = c + 16*lane  ->  phase v=c (q) / w=c (k), m=lane
#pragma unroll
    for (int c = 0; c < 16; ++c)
      qsh[wave][c * 34 + lane] = (unsigned short)f2h(qv[c]);   // qsh[v][m]
    uint32_t* kdst = (uint32_t*)&ksh[wave][lane * 18];         // ksh[m][w], packed pairs
#pragma unroll
    for (int j = 0; j < 8; ++j)
      kdst[j] = f2h(kv[2*j]) | (f2h(kv[2*j+1]) << 16);
  }
#pragma unroll
  for (int i = 0; i < 16; ++i) corr[wave][i * 32 + lane] = 0.0f;

  __syncthreads();

  // ---------------- A fragment (q phases), fixed across all 32 lags --------
  // 16-bit A 16x32 layout: lane&15 = M(=v); VGPR j holds K pair
  // m0 = (j>>2)*16 + (lane>>4)*8 + (j&3)*2
  const int hi   = lane >> 4;
  const int vIdx = lane & 15;
  union { v16h h; uint32_t u[8]; } A;
  {
    const uint32_t* qb = (const uint32_t*)&qsh[wave][vIdx * 34];
#pragma unroll
    for (int j = 0; j < 8; ++j) {
      int m0 = ((j >> 2) << 4) + (hi << 3) + ((j & 3) << 1);
      A.u[j] = qb[m0 >> 1];
    }
  }

  // scatter targets: D[v][w] -> corr[((v-w)&15) + 16*((l-(w>v))&31)]
  const int w = lane & 15;          // C/D layout: N = lane&15, M = j + 8*(lane>>4)
  int sArr[8], cArr[8];
#pragma unroll
  for (int j = 0; j < 8; ++j) {
    int vv  = j + (hi << 3);
    sArr[j] = (vv - w) & 15;
    cArr[j] = (w > vv) ? 1 : 0;
  }

  // ---------------- 32 lags: one V_WMMA_F32_16X16X32_F16 each --------------
#pragma unroll 8
  for (int l = 0; l < 32; ++l) {
    // B 32x16: lane&15 = N (w); VGPR j holds K pair m0 (same as A). lag = K-row rotation.
    union { v16h h; uint32_t u[8]; } B;
    const unsigned short* kb = &ksh[wave][w];
#pragma unroll
    for (int j = 0; j < 8; ++j) {
      int k0 = ((j >> 2) << 4) + (hi << 3) + ((j & 3) << 1);
      B.u[j] = (uint32_t)kb[((k0 - l) & 31) * 18] | ((uint32_t)kb[((k0 + 1 - l) & 31) * 18] << 16);
    }

    v8f acc = {};
    acc = __builtin_amdgcn_wmma_f32_16x16x32_f16(false, A.h, false, B.h,
                                                 (short)0, acc, false, false);
#pragma unroll
    for (int j = 0; j < 8; ++j) {
      int t = sArr[j] + (((l - cArr[j]) & 31) << 4);
      atomicAdd(&corr[wave][t], acc[j]);   // ds_add_f32
    }
  }

  __syncthreads();

  // ---------------- top-7 mean over this row's 512 corr values -------------
  float arr[16];
#pragma unroll
  for (int i = 0; i < 16; ++i) arr[i] = corr[wave][lane * 16 + i];
  float sum = 0.0f;
#pragma unroll 1
  for (int it = 0; it < 7; ++it) {
    float loc = -3.402823e38f; int li = 0;
#pragma unroll
    for (int i = 0; i < 16; ++i) if (arr[i] > loc) { loc = arr[i]; li = i; }
    float m = loc;
#pragma unroll
    for (int off = 16; off > 0; off >>= 1) m = fmaxf(m, __shfl_xor(m, off, 32));
    sum += m;
    int cand = (loc == m) ? lane : 32;          // tie-break: lowest lane removes
#pragma unroll
    for (int off = 16; off > 0; off >>= 1) cand = min(cand, __shfl_xor(cand, off, 32));
    const bool rm = (lane == cand);
#pragma unroll
    for (int i = 0; i < 16; ++i) arr[i] = (rm && i == li) ? -3.402823e38f : arr[i];
  }
  const float meanv = sum * (1.0f / 7.0f);

  // ---------------- blend: out = v*sig(corr-mean) + roll(v)*(1-sig) --------
  const float* vrow = V + (size_t)row * DM;
  const float* vnxt = V + (size_t)rowN * DM;
  float*       orow = Out + (size_t)row * DM;
#pragma unroll
  for (int i = 0; i < 16; ++i) {
    int d = lane + (i << 5);                    // coalesced
    float cc = corr[wave][d];
    float w0 = 1.0f / (1.0f + __expf(meanv - cc));  // softmax([c,m]) pair
    orow[d] = vrow[d] * w0 + vnxt[d] * (1.0f - w0);
  }
}

extern "C" void kernel_launch(void* const* d_in, const int* in_sizes, int n_in,
                              void* d_out, int out_size, void* d_ws, size_t ws_size,
                              hipStream_t stream) {
  const float* Q = (const float*)d_in[0];
  const float* K = (const float*)d_in[1];
  const float* V = (const float*)d_in[2];
  float* Out = (float*)d_out;
  const int rows   = in_sizes[0] / DM;   // 32*2048 = 65536
  const int blocks = rows / WPB;         // 8192
  autocorr_wmma_kernel<<<blocks, 256, 0, stream>>>(Q, K, V, Out);
}
